// AdaptivePCEN_40381282517134
// MI455X (gfx1250) — matrix operand order repaired and checked by reading, not compiled, not run on hardware
//
#include <hip/hip_runtime.h>
#include <hip/hip_bf16.h>

typedef __attribute__((ext_vector_type(16))) _Float16 v16h;
typedef __attribute__((ext_vector_type(8)))  _Float16 v8h;
typedef __attribute__((ext_vector_type(8)))  float    v8f;

#define B_  32
#define F_  128
#define T_  4000
#define H_  256
#define TWO_F 256
#define OUT4  512
#define NT_TILES (T_ / 16)   // 250

#define CAT16(lo, hi) __builtin_shufflevector(lo, hi, 0,1,2,3,4,5,6,7,8,9,10,11,12,13,14,15)

__device__ __forceinline__ float sigmoidf(float x) {
    return 1.0f / (1.0f + __expf(-x));
}
__device__ __forceinline__ float softplusf(float x) {
    return fmaxf(x, 0.0f) + log1pf(__expf(-fabsf(x)));   // stable
}

// ---------------------------------------------------------------------------
// Kernel 0: one-time weight conversion to f16, K-contiguous per column.
//   W1t[n][k] = (f16) W1[k][n]   (n<256, k<256)
//   W2t[n][k] = (f16) W2[k][n]   (n<512, k<256)
// ---------------------------------------------------------------------------
__global__ __launch_bounds__(256) void k_prep(const float* __restrict__ W1,
                                              const float* __restrict__ W2,
                                              _Float16* __restrict__ W1t,
                                              _Float16* __restrict__ W2t) {
    const int i = blockIdx.x * 256 + threadIdx.x;
    if (i < H_ * TWO_F) {                       // 65536
        const int n = i >> 8, k = i & 255;
        W1t[i] = (_Float16)W1[(size_t)k * H_ + n];
    }
    const int j = i - H_ * TWO_F;
    if (j >= 0 && j < OUT4 * H_) {              // 131072
        const int n = j >> 8, k = j & 255;
        W2t[j] = (_Float16)W2[(size_t)k * OUT4 + n];
    }
}

// ---------------------------------------------------------------------------
// Kernel 1: H = relu( concat(X_prev, X)^T @ W1 + b1 )  -> f16 [B, T, H]
// One block = one (b, 16-frame t-tile). 8 waves; each wave owns 2 N-tiles.
// ---------------------------------------------------------------------------
__global__ __launch_bounds__(256) void k_gemm1(const float* __restrict__ X,
                                               const _Float16* __restrict__ W1t,
                                               const float* __restrict__ b1,
                                               _Float16* __restrict__ Hbuf) {
    __shared__ _Float16 As[16][TWO_F];   // [M][K], 8 KB

    const int bt   = blockIdx.x;
    const int b    = bt / NT_TILES;
    const int t0   = (bt % NT_TILES) * 16;
    const int tid  = threadIdx.x;
    const int lane = tid & 31;
    const int wave = tid >> 5;
    const int half = lane >> 4;
    const int ln   = lane & 15;

    // ---- stage A tile: As[m][k] = concat(X_prev,X)[b, k, t0+m] (k = tid) ----
    {
        const int k = tid;
        if (k < F_) {                       // X_prev half: X[b,k,t-1], clamped
            const float* src = X + ((size_t)b * F_ + k) * T_;
            #pragma unroll
            for (int m = 0; m < 16; ++m) {
                int t = t0 + m - 1;
                t = (t < 0) ? 0 : t;
                As[m][k] = (_Float16)src[t];
            }
        } else {                            // current half: X[b,k-F,t]
            const float* src = X + ((size_t)b * F_ + (k - F_)) * T_;
            #pragma unroll
            for (int m = 0; m < 16; ++m)
                As[m][k] = (_Float16)src[t0 + m];
        }
    }
    __syncthreads();

    const int n0 = wave * 32 + ln;          // N-tile 0 column
    const int n1 = n0 + 16;                 // N-tile 1 column
    const _Float16* w0row = W1t + (size_t)n0 * TWO_F;
    const _Float16* w1row = W1t + (size_t)n1 * TWO_F;

    v8f acc0 = {}, acc1 = {};
    #pragma unroll
    for (int ks = 0; ks < 8; ++ks) {        // K = 256 in steps of 32
        const int k0 = ks * 32;
        // A fragment: halfwords 0..15 = K k0+half*16..k0+half*16+15
        const v8h alo = *(const v8h*)&As[ln][k0 + half * 16];
        const v8h ahi = *(const v8h*)&As[ln][k0 + half * 16 + 8];
        const v16h a = CAT16(alo, ahi);
        // B fragment: lane-half picks 16-consecutive-K block
        const v8h b0lo = *(const v8h*)(w0row + k0 + half * 16);
        const v8h b0hi = *(const v8h*)(w0row + k0 + half * 16 + 8);
        const v8h b1lo = *(const v8h*)(w1row + k0 + half * 16);
        const v8h b1hi = *(const v8h*)(w1row + k0 + half * 16 + 8);
        acc0 = __builtin_amdgcn_wmma_f32_16x16x32_f16(
                   false, a, false, CAT16(b0lo, b0hi), (short)0, acc0, false, false);
        acc1 = __builtin_amdgcn_wmma_f32_16x16x32_f16(
                   false, a, false, CAT16(b1lo, b1hi), (short)0, acc1, false, false);
    }

    // epilogue: bias + relu, store f16  (C/D: VGPR r -> M = r + 8*half)
    const float bias0 = b1[n0];
    const float bias1 = b1[n1];
    #pragma unroll
    for (int r = 0; r < 8; ++r) {
        const int m = r + half * 8;
        float v0 = acc0[r] + bias0;
        float v1 = acc1[r] + bias1;
        v0 = v0 > 0.0f ? v0 : 0.0f;
        v1 = v1 > 0.0f ? v1 : 0.0f;
        _Float16* hrow = Hbuf + ((size_t)b * T_ + (t0 + m)) * H_;
        hrow[n0] = (_Float16)v0;
        hrow[n1] = (_Float16)v1;
    }
}

// ---------------------------------------------------------------------------
// Kernel 2: OUT = H @ W2 + b2 ; activations ; scatter to SAD[4][B][F][T] (f32)
// One block = one (b, t-tile). 8 waves x 4 N-tiles cover N = 512.
// ---------------------------------------------------------------------------
__global__ __launch_bounds__(256) void k_gemm2(const _Float16* __restrict__ Hbuf,
                                               const _Float16* __restrict__ W2t,
                                               const float* __restrict__ b2,
                                               float* __restrict__ SAD) {
    const int bt   = blockIdx.x;
    const int b    = bt / NT_TILES;
    const int t0   = (bt % NT_TILES) * 16;
    const int tid  = threadIdx.x;
    const int lane = tid & 31;
    const int wave = tid >> 5;
    const int half = lane >> 4;
    const int ln   = lane & 15;

    const _Float16* arow = Hbuf + ((size_t)b * T_ + (t0 + ln)) * H_;

    v8f acc[4] = {{}, {}, {}, {}};
    #pragma unroll
    for (int ks = 0; ks < 8; ++ks) {            // K = 256 in steps of 32
        const int k0 = ks * 32;
        const v8h alo = *(const v8h*)(arow + k0 + half * 16);
        const v8h ahi = *(const v8h*)(arow + k0 + half * 16 + 8);
        const v16h a = CAT16(alo, ahi);
        #pragma unroll
        for (int nt = 0; nt < 4; ++nt) {
            const int n = wave * 64 + nt * 16 + ln;     // 0..511
            const _Float16* wrow = W2t + (size_t)n * H_;
            const v8h wlo = *(const v8h*)(wrow + k0 + half * 16);
            const v8h whi = *(const v8h*)(wrow + k0 + half * 16 + 8);
            acc[nt] = __builtin_amdgcn_wmma_f32_16x16x32_f16(
                          false, a, false, CAT16(wlo, whi), (short)0, acc[nt], false, false);
        }
    }

    // epilogue: bias + activation, vectorized write to [q][B][F][T]
    #pragma unroll
    for (int nt = 0; nt < 4; ++nt) {
        const int o  = wave * 64 + nt * 16 + ln;   // 0..511
        const int q  = o >> 7;                     // 0=s 1=alpha 2=delta 3=r
        const int f  = o & 127;
        const float bias = b2[o];
        float tmp[8];
        #pragma unroll
        for (int r = 0; r < 8; ++r) {
            float v = acc[nt][r] + bias;
            tmp[r] = (q == 2) ? softplusf(v) : sigmoidf(v);
        }
        // this lane owns 8 consecutive t values starting at t0 + half*8
        float* dst = SAD + (((size_t)q * B_ + b) * F_ + f) * T_ + t0 + half * 8;
        *(float4*)(dst + 0) = make_float4(tmp[0], tmp[1], tmp[2], tmp[3]);
        *(float4*)(dst + 4) = make_float4(tmp[4], tmp[5], tmp[6], tmp[7]);
    }
}

// ---------------------------------------------------------------------------
// Kernel 3: EMA scan over t + PCEN normalization. One thread per (b,f).
// ---------------------------------------------------------------------------
__global__ __launch_bounds__(256) void k_scan(const float* __restrict__ X,
                                              const float* __restrict__ SAD,
                                              float* __restrict__ out) {
    const int ch = blockIdx.x * blockDim.x + threadIdx.x;   // 0 .. B*F-1
    if (ch >= B_ * F_) return;

    const size_t CT = (size_t)B_ * F_ * T_;
    const float* s  = SAD + 0 * CT + (size_t)ch * T_;
    const float* al = SAD + 1 * CT + (size_t)ch * T_;
    const float* de = SAD + 2 * CT + (size_t)ch * T_;
    const float* rr = SAD + 3 * CT + (size_t)ch * T_;
    const float* x  = X + (size_t)ch * T_;
    float* o        = out + (size_t)ch * T_;

    float M = 0.0f;
    for (int t = 0; t < T_; ++t) {
        if ((t & 15) == 0 && t + 64 < T_) {
            __builtin_prefetch(&s[t + 64], 0, 0);
            __builtin_prefetch(&x[t + 64], 0, 0);
            __builtin_prefetch(&al[t + 64], 0, 0);
            __builtin_prefetch(&de[t + 64], 0, 0);
            __builtin_prefetch(&rr[t + 64], 0, 0);
        }
        const float st = s[t];
        const float xt = x[t];
        M = (1.0f - st) * M + st * xt;
        const float a = al[t], d = de[t], r = rr[t];
        const float Ma = __powf(M + 1e-6f, a);
        o[t] = __powf(xt / Ma + d, r) - __powf(d, r);
    }
}

extern "C" void kernel_launch(void* const* d_in, const int* in_sizes, int n_in,
                              void* d_out, int out_size, void* d_ws, size_t ws_size,
                              hipStream_t stream) {
    const float* X  = (const float*)d_in[0];   // [B,F,T]
    const float* W1 = (const float*)d_in[1];   // [2F,H]
    const float* b1 = (const float*)d_in[2];   // [H]
    const float* W2 = (const float*)d_in[3];   // [H,4F]
    const float* b2 = (const float*)d_in[4];   // [4F]
    float* out = (float*)d_out;                // [B,F,T]

    // workspace layout (16B aligned): W1t f16 | W2t f16 | Hbuf f16 | SAD f32
    char* ws = (char*)d_ws;
    _Float16* W1t = (_Float16*)ws;                                   // 128 KB
    _Float16* W2t = (_Float16*)(ws + (size_t)H_ * TWO_F * 2);        // 256 KB
    _Float16* Hbuf = (_Float16*)(ws + (size_t)(H_ * TWO_F + OUT4 * H_) * 2);
    float* SAD = (float*)((char*)Hbuf + (size_t)B_ * T_ * H_ * sizeof(_Float16));

    k_prep<<<(H_ * TWO_F + OUT4 * H_ + 255) / 256, 256, 0, stream>>>(W1, W2, W1t, W2t);
    const int nblocks = B_ * NT_TILES;   // 8000
    k_gemm1<<<nblocks, 256, 0, stream>>>(X, W1t, b1, Hbuf);
    k_gemm2<<<nblocks, 256, 0, stream>>>(Hbuf, W2t, b2, SAD);
    k_scan<<<(B_ * F_ + 255) / 256, 256, 0, stream>>>(X, SAD, out);
}
